// NCA_18940805775935
// MI455X (gfx1250) — matrix operand order repaired and checked by reading, not compiled, not run on hardware
//
#include <hip/hip_runtime.h>
#include <hip/hip_bf16.h>

typedef __attribute__((ext_vector_type(16))) _Float16 v16h;
typedef __attribute__((ext_vector_type(8)))  _Float16 v8h;
typedef __attribute__((ext_vector_type(8)))  float    v8f;

#define NCA_B    16
#define NCA_H    256
#define NCA_C    14
#define NCA_PERC 51      // (C+3)*3 perception features
#define NCA_K1   64      // padded K for GEMM1
#define NCA_HID  256
#define WAVES    4       // waves per block
#define PXB      64      // pixels per block (WAVES * 16)
#define WINW     66      // window width incl. halo

// ---------------- gfx1250 async global->LDS copy (ASYNCcnt path) ------------
#if defined(__gfx1250__) && __has_builtin(__builtin_amdgcn_global_load_async_to_lds_b64) && \
    __has_builtin(__builtin_amdgcn_global_load_async_to_lds_b32)
#define NCA_ASYNC 1
typedef int nca_v2i __attribute__((vector_size(2 * sizeof(int))));
#define NCA_AS1 __attribute__((address_space(1)))
#define NCA_AS3 __attribute__((address_space(3)))
#endif

static __device__ inline void async_copy_b64(const float* g, float* l) {
#if defined(NCA_ASYNC)
  __builtin_amdgcn_global_load_async_to_lds_b64(
      (NCA_AS1 nca_v2i*)g, (NCA_AS3 nca_v2i*)l, 0, 0);
#else
  *(double*)l = *(const double*)g;
#endif
}
static __device__ inline void async_copy_b32(const float* g, float* l) {
#if defined(NCA_ASYNC)
  __builtin_amdgcn_global_load_async_to_lds_b32(
      (NCA_AS1 int*)g, (NCA_AS3 int*)l, 0, 0);
#else
  *l = *g;
#endif
}
static __device__ inline void async_wait0() {
#if defined(NCA_ASYNC)
#if __has_builtin(__builtin_amdgcn_s_wait_asynccnt)
  __builtin_amdgcn_s_wait_asynccnt(0);
#else
  asm volatile("s_wait_asynccnt 0" ::: "memory");
#endif
#endif
}

static __device__ inline v8f wmma_f16(v16h a, v16h b, v8f c) {
  // D = A(16x32) * B(32x16) + C  -> v_wmma_f32_16x16x32_f16
  return __builtin_amdgcn_wmma_f32_16x16x32_f16(
      false, a, false, b, (short)0, c, false, false);
}

static __device__ inline float hash_uniform(unsigned v) {
  v ^= v >> 17; v *= 0xed5ad4bbu;
  v ^= v >> 11; v *= 0xac4c1b51u;
  v ^= v >> 15; v *= 0x31848babu;
  v ^= v >> 14;
  return (float)(v >> 8) * (1.0f / 16777216.0f);
}

// ---------------------------------------------------------------------------
// Kernel 1: fused perception -> GEMM1(51x256)+ReLU -> GEMM2(256x14) -> update
// Block = 128 threads (4 waves) = 64 contiguous pixels of one image row.
// Halo window staged in LDS via async global->LDS; GEMMs via f16 WMMA.
// Writes UNMASKED x_new into d_out (masking done by kernels 2/3).
// ---------------------------------------------------------------------------
__global__ __launch_bounds__(128) void nca_fused_kernel(
    const float* __restrict__ xin, const float* __restrict__ img,
    const float* __restrict__ W1,  const float* __restrict__ b1,
    const float* __restrict__ W2,  const float* __restrict__ b2,
    float* __restrict__ out) {

  __shared__ __attribute__((aligned(128))) _Float16 W1t[NCA_HID][NCA_K1];   // 32 KB
  __shared__ __attribute__((aligned(128))) _Float16 W2t[16][NCA_HID];       //  8 KB
  __shared__ __attribute__((aligned(128))) _Float16 hstage[WAVES][16][32];  //  4 KB
  __shared__ __attribute__((aligned(16)))  float    xwin[3][WINW][NCA_C];   // 11.1 KB
  __shared__ __attribute__((aligned(16)))  float    imgw[3][WINW][3];       //  2.4 KB

  const int tid = threadIdx.x;           // 0..127
  const int wg  = blockIdx.x;            // 16 * 256 * 4
  const int px0 = (wg & 3) * PXB;
  const int y   = (wg >> 2) & (NCA_H - 1);
  const int b   = wg >> 10;

  // ---- weights: f32 -> f16, transposed so B operands are lane-contiguous
#pragma unroll 4
  for (int i = tid; i < NCA_HID * NCA_K1; i += 128) {
    int n = i >> 6, k = i & 63;
    W1t[n][k] = (_Float16)((k < NCA_PERC) ? W1[k * NCA_HID + n] : 0.0f);
  }
#pragma unroll 4
  for (int i = tid; i < 16 * NCA_HID; i += 128) {
    int n = i >> 8, k = i & 255;
    W2t[n][k] = (_Float16)((n < NCA_C) ? W2[k * NCA_C + n] : 0.0f);
  }

  // ---- stage halo window: 3 rows x [px0-1, px0+64] into LDS (async)
  const int gx0 = px0 - 1;
  const int gxs = (gx0 < 0) ? 0 : gx0;                       // first valid gx
  const int gxe = (px0 + PXB > NCA_H - 1) ? (NCA_H - 1) : (px0 + PXB);
  const int npx = gxe - gxs + 1;                             // 65 or 66
  const int lxs = gxs - gx0;                                 // 0 or 1

  for (int r = 0; r < 3; ++r) {
    const int yy = y + r - 1;
    if ((unsigned)yy < (unsigned)NCA_H) {
      const float* gx_ = xin + ((size_t)((b * NCA_H + yy) * NCA_H + gxs)) * NCA_C;
      float* lx_ = &xwin[r][lxs][0];
      const int nch = npx * (NCA_C / 2);                     // b64 chunks
      for (int i = tid; i < nch; i += 128)
        async_copy_b64(gx_ + i * 2, lx_ + i * 2);
      const float* gi_ = img + ((size_t)(yy * NCA_H + gxs)) * 3;
      float* li_ = &imgw[r][lxs][0];
      const int nci = npx * 3;                               // b32 chunks
      for (int i = tid; i < nci; i += 128)
        async_copy_b32(gi_ + i, li_ + i);
    } else {
      float* zx = &xwin[r][0][0];
      for (int i = tid; i < WINW * NCA_C; i += 128) zx[i] = 0.0f;
      float* zi = &imgw[r][0][0];
      for (int i = tid; i < WINW * 3; i += 128) zi[i] = 0.0f;
    }
  }
  // zero out-of-image halo columns
  if (tid < 3) {
    const int r = tid;
    if (gx0 < 0) {
#pragma unroll
      for (int c = 0; c < NCA_C; ++c) xwin[r][0][c] = 0.0f;
#pragma unroll
      for (int c = 0; c < 3; ++c) imgw[r][0][c] = 0.0f;
    }
    if (px0 + PXB > NCA_H - 1) {
#pragma unroll
      for (int c = 0; c < NCA_C; ++c) xwin[r][WINW - 1][c] = 0.0f;
#pragma unroll
      for (int c = 0; c < 3; ++c) imgw[r][WINW - 1][c] = 0.0f;
    }
  }
  async_wait0();
  __syncthreads();

  // ---- WMMA role decomposition (wave32)
  const int lane = tid & 31;
  const int wave = tid >> 5;
  const int m    = lane & 15;            // A row / D column owner
  const int n0   = lane & 15;
  const int half = lane >> 4;
  const int o    = half * 8;
  const int lx   = wave * 16 + m + 1;    // window x of this lane's pixel

  // concatenated-channel window read (y/x OOB already zeroed in LDS)
  auto wval = [&](int r, int xx, int c) -> float {
    return (c < NCA_C) ? xwin[r][xx][c] : imgw[r][xx][c - NCA_C];
  };
  // perception feature k = c*3+f : identity / sobel-x / sobel-y
  auto perc = [&](int c, int f) -> float {
    if (f == 0) return wval(1, lx, c);
    float s = 0.0f;
    if (f == 1) {
#pragma unroll
      for (int dy = 0; dy < 3; ++dy) {
        float rw = (dy == 1) ? 2.0f : 1.0f;
        s += rw * (wval(dy, lx + 1, c) - wval(dy, lx - 1, c));
      }
    } else {
#pragma unroll
      for (int dxx = -1; dxx <= 1; ++dxx) {
        float cw = (dxx == 0) ? 2.0f : 1.0f;
        s += cw * (wval(2, lx + dxx, c) - wval(0, lx + dxx, c));
      }
    }
    return s * 0.125f;
  };

  // ---- build GEMM1 A operands (16-bit A 16x32 layout)
  v16h a0, a1;
#pragma unroll
  for (int e = 0; e < 16; ++e) {
    const int kk = half * 16 + e;
    a0[e] = (_Float16)perc(kk / 3, kk % 3);                  // K 0..31
    const int k1i = 32 + kk;
    a1[e] = (_Float16)((k1i < NCA_PERC) ? perc(k1i / 3, k1i % 3) : 0.0f);
  }

  // ---- GEMM2 accumulator init with b2
  const float b2v = (n0 < NCA_C) ? b2[n0] : 0.0f;
  v8f c2;
#pragma unroll
  for (int r = 0; r < 8; ++r) c2[r] = b2v;

  // ---- main loop: pairs of hidden N-tiles feed one GEMM2 k-step (K=32)
#pragma unroll 2
  for (int ntp = 0; ntp < 8; ++ntp) {
#pragma unroll
    for (int sub = 0; sub < 2; ++sub) {
      const int nt   = ntp * 2 + sub;
      const int ncol = nt * 16 + n0;
      const float b1v = b1[ncol];
      v8f c;
#pragma unroll
      for (int r = 0; r < 8; ++r) c[r] = b1v;

      v16h bB0 = *(const v16h*)&W1t[ncol][half * 16];
      v16h bB1 = *(const v16h*)&W1t[ncol][32 + half * 16];
      c = wmma_f16(a0, bB0, c);
      c = wmma_f16(a1, bB1, c);

#pragma unroll
      for (int r = 0; r < 8; ++r) {
        float v = c[r] > 0.0f ? c[r] : 0.0f;                 // ReLU
        hstage[wave][half * 8 + r][sub * 16 + n0] = (_Float16)v;
      }
    }
    // staged h -> GEMM2 A operand (row m, 32 K-values), wave-private LDS
    v8h lo = *(const v8h*)&hstage[wave][m][half * 16];
    v8h hi = *(const v8h*)&hstage[wave][m][half * 16 + 8];
    v16h a2;
#pragma unroll
    for (int e = 0; e < 8; ++e) { a2[e] = lo[e]; a2[e + 8] = hi[e]; }

    v16h bW2 = *(const v16h*)&W2t[n0][ntp * 32 + half * 16];
    c2 = wmma_f16(a2, bW2, c2);
  }

  // ---- epilogue: x_new = x + 0.1 * dx * bernoulli(0.5)  (unmasked)
  const float paso = 0.1f;
#pragma unroll
  for (int r = 0; r < 8; ++r) {
    const int M  = half * 8 + r;
    const int pp = px0 + wave * 16 + M;
    const int ch = n0;
    if (ch < NCA_C) {
      const unsigned pid = (unsigned)((b * NCA_H + y) * NCA_H + pp);
      const float u = hash_uniform(pid * 0x9E3779B9u + 0x85EBCA6Bu);
      const float updf = (u <= 0.5f) ? 1.0f : 0.0f;
      const size_t idx = (size_t)pid * NCA_C + ch;
      out[idx] = xin[idx] + c2[r] * paso * updf;
    }
  }
}

// ---------------------------------------------------------------------------
// Kernel 2: alive mask = (maxpool3x3(x_old ch0)>.1) && (maxpool3x3(x_new ch0)>.1)
// ---------------------------------------------------------------------------
__global__ __launch_bounds__(256) void nca_mask_kernel(
    const float* __restrict__ xin, const float* __restrict__ xnew,
    unsigned char* __restrict__ mask) {
  const int pid = blockIdx.x * 256 + threadIdx.x;
  const int bb = pid >> 16;
  const int y  = (pid >> 8) & 255;
  const int x  = pid & 255;
  float m0 = -1e30f, m1 = -1e30f;
#pragma unroll
  for (int dy = -1; dy <= 1; ++dy)
#pragma unroll
    for (int dx = -1; dx <= 1; ++dx) {
      const int yy = y + dy, xx = x + dx;
      if ((unsigned)yy < (unsigned)NCA_H && (unsigned)xx < (unsigned)NCA_H) {
        const size_t q = ((size_t)((bb * NCA_H + yy) * NCA_H + xx)) * NCA_C;
        m0 = fmaxf(m0, xin[q]);
        m1 = fmaxf(m1, xnew[q]);
      }
    }
  mask[pid] = (m0 > 0.1f && m1 > 0.1f) ? 1 : 0;
}

// ---------------------------------------------------------------------------
// Kernel 3: apply mask across 14 channels
// ---------------------------------------------------------------------------
__global__ __launch_bounds__(256) void nca_apply_kernel(
    float* __restrict__ xnew, const unsigned char* __restrict__ mask) {
  const int pid = blockIdx.x * 256 + threadIdx.x;
  const float s = mask[pid] ? 1.0f : 0.0f;
  const size_t q = (size_t)pid * NCA_C;
#pragma unroll
  for (int ch = 0; ch < NCA_C; ++ch) xnew[q + ch] *= s;
}

extern "C" void kernel_launch(void* const* d_in, const int* in_sizes, int n_in,
                              void* d_out, int out_size, void* d_ws, size_t ws_size,
                              hipStream_t stream) {
  (void)in_sizes; (void)n_in; (void)out_size; (void)ws_size;
  const float* x   = (const float*)d_in[0];
  const float* img = (const float*)d_in[1];
  const float* W1  = (const float*)d_in[2];
  const float* b1  = (const float*)d_in[3];
  const float* W2  = (const float*)d_in[4];
  const float* b2  = (const float*)d_in[5];
  float* out = (float*)d_out;
  unsigned char* mask = (unsigned char*)d_ws;           // 1 MB of scratch

  const int n_pixels = NCA_B * NCA_H * NCA_H;           // 1,048,576
  const int grid1 = NCA_B * NCA_H * (NCA_H / PXB);      // 16384
  const int grid2 = n_pixels / 256;                     // 4096

  nca_fused_kernel<<<grid1, 128, 0, stream>>>(x, img, W1, b1, W2, b2, out);
  nca_mask_kernel<<<grid2, 256, 0, stream>>>(x, out, mask);
  nca_apply_kernel<<<grid2, 256, 0, stream>>>(out, mask);
}
